// Qwen3MoeSparseMoeBlock_10187662426953
// MI455X (gfx1250) — compile-verified, matrix operand order checked by reading, not run on hardware
//
#include <hip/hip_runtime.h>
#include <hip/hip_bf16.h>

// ---------------- problem constants ----------------
constexpr int Bc = 2, Sc = 1024, Hc = 2048;
constexpr int Tc = Bc * Sc;          // 2048 tokens
constexpr int Ec = 64, Kc = 8, Fc = 768, CAPc = 512;

// ---------------- WMMA types ----------------
typedef __attribute__((ext_vector_type(16))) __bf16 v16bf;
typedef __attribute__((ext_vector_type(8)))  float  v8f;

union Frag16 { v16bf v; uint4 q[2]; };
static_assert(sizeof(Frag16) == 32, "frag size");

// Load one 16x32 bf16 A/B fragment from an LDS tile laid out [row][k], row
// stride = lda (bf16 elements, multiple of 8 so 16B aligned).
// Per ISA: lanes 0-15 hold K = 0..7 and 16..23; lanes 16-31 hold K = 8..15
// and 24..31 -> two contiguous 16-byte runs per lane.
__device__ __forceinline__ v16bf load_frag(const __hip_bfloat16* base, int row0,
                                           int lda, int lane) {
  const int r  = row0 + (lane & 15);
  const int k0 = ((lane >> 4) & 1) * 8;
  Frag16 f;
  f.q[0] = *(const uint4*)(base + r * lda + k0);
  f.q[1] = *(const uint4*)(base + r * lda + k0 + 16);
  return f.v;
}

#define WMMA_BF16(A, B, C) \
  __builtin_amdgcn_wmma_f32_16x16x32_bf16(false, (A), false, (B), (short)0, (C), false, false)

// pack two fp32 -> packed bf16x2 dword (for b32 LDS stores)
__device__ __forceinline__ unsigned pack2bf(float a, float b) {
  union { __hip_bfloat16 h[2]; unsigned u; } p;
  p.h[0] = __float2bfloat16(a);
  p.h[1] = __float2bfloat16(b);
  return p.u;
}

// Async global->LDS copy of 16 bytes (bf16 data, no conversion needed).
// Tracked by ASYNCcnt; caller must issue s_wait_asynccnt 0 before the barrier.
__device__ __forceinline__ void async_copy_b128(const __hip_bfloat16* gsrc,
                                                __hip_bfloat16* lds_dst) {
  const unsigned lds_off = (unsigned)(uintptr_t)lds_dst;         // LDS aperture: addr[31:0]
  const unsigned long long ga = (unsigned long long)(uintptr_t)gsrc;
  asm volatile("global_load_async_to_lds_b128 %0, %1, off"
               :: "v"(lds_off), "v"(ga)
               : "memory");
}
__device__ __forceinline__ void wait_async0() {
  asm volatile("s_wait_asynccnt 0" ::: "memory");
}

// =====================================================================
// 1) Router: logits, softmax, top-8, renormalize
// =====================================================================
__global__ __launch_bounds__(64)
void moe_router(const float* __restrict__ x, const float* __restrict__ rw,
                int* __restrict__ topk_idx, float* __restrict__ topk_w) {
  const int t = blockIdx.x;
  __shared__ __align__(16) float xs[Hc];
  __shared__ float lg[Ec];
  __shared__ float sc[Ec];

  const int tid = threadIdx.x;
  const float4* xsrc = (const float4*)(x + (size_t)t * Hc);
  float4* xdst = (float4*)xs;
  for (int i = tid; i < Hc / 4; i += 64) xdst[i] = xsrc[i];
  __syncthreads();

  {
    const float* w = rw + (size_t)tid * Hc;
    float acc = 0.f;
    for (int h = 0; h < Hc; h += 4) {
      float4 wv = *(const float4*)(w + h);
      acc += xs[h + 0] * wv.x + xs[h + 1] * wv.y + xs[h + 2] * wv.z + xs[h + 3] * wv.w;
    }
    lg[tid] = acc;
  }
  __syncthreads();

  if (tid == 0) {
    float mx = lg[0];
    for (int e = 1; e < Ec; ++e) mx = fmaxf(mx, lg[e]);
    float se = 0.f;
    for (int e = 0; e < Ec; ++e) { float ex = __expf(lg[e] - mx); sc[e] = ex; se += ex; }
    const float invse = 1.f / se;
    for (int e = 0; e < Ec; ++e) sc[e] *= invse;          // softmax scores

    int   idxs[Kc];
    float ws8[Kc];
    float sum8 = 0.f;
    for (int k = 0; k < Kc; ++k) {
      float best = -1.f; int bi = 0;
      for (int e = 0; e < Ec; ++e) { float v = sc[e]; if (v > best) { best = v; bi = e; } }
      sc[bi] = -1.f;
      idxs[k] = bi; ws8[k] = best; sum8 += best;
    }
    const float inv = 1.f / (sum8 + 1e-20f);
    for (int k = 0; k < Kc; ++k) {
      topk_idx[t * Kc + k] = idxs[k];
      topk_w[t * Kc + k]   = ws8[k] * inv;
    }
  }
}

// =====================================================================
// 2) Dispatch: sequential per-expert position assignment (matches the
//    reference cumsum-in-slot-order semantics exactly).
// =====================================================================
__global__ __launch_bounds__(64)
void moe_dispatch(const int* __restrict__ topk_idx, int* __restrict__ dest,
                  int* __restrict__ token_of, int* __restrict__ counts) {
  const int e = threadIdx.x;
  int cnt = 0;
  for (int s = 0; s < Tc * Kc; ++s) {
    if (topk_idx[s] == e) {
      if (cnt < CAPc) {
        dest[s] = e * CAPc + cnt;
        token_of[e * CAPc + cnt] = s >> 3;   // s / K
      } else {
        dest[s] = Ec * CAPc;                 // dropped
      }
      ++cnt;
    }
  }
  counts[e] = cnt < CAPc ? cnt : CAPc;
}

// =====================================================================
// 3) Gather: build bf16 capacity buffer xb [E*CAP, H] (zeros where unused)
// =====================================================================
__global__ __launch_bounds__(256)
void moe_gather(const float* __restrict__ x, const int* __restrict__ token_of,
                const int* __restrict__ counts, __hip_bfloat16* __restrict__ xb) {
  const int row = blockIdx.x;                  // 0..E*CAP-1
  const int e   = row >> 9;                    // /CAP
  const int pos = row & (CAPc - 1);
  const int c0  = threadIdx.x * 8;

  union { __hip_bfloat16 h[8]; uint4 q; } pk;
  if (pos < counts[e]) {
    const int tok = token_of[row];
    const float4 f0 = *(const float4*)(x + (size_t)tok * Hc + c0);
    const float4 f1 = *(const float4*)(x + (size_t)tok * Hc + c0 + 4);
    pk.h[0] = __float2bfloat16(f0.x); pk.h[1] = __float2bfloat16(f0.y);
    pk.h[2] = __float2bfloat16(f0.z); pk.h[3] = __float2bfloat16(f0.w);
    pk.h[4] = __float2bfloat16(f1.x); pk.h[5] = __float2bfloat16(f1.y);
    pk.h[6] = __float2bfloat16(f1.z); pk.h[7] = __float2bfloat16(f1.w);
  } else {
    for (int j = 0; j < 8; ++j) pk.h[j] = __float2bfloat16(0.f);
  }
  *(uint4*)(xb + (size_t)row * Hc + c0) = pk.q;
}

// =====================================================================
// 4) GEMM1: hmid = silu(xb @ w_gate) * (xb @ w_up)   per expert
//    block tile 128x64, BK=32, 8 waves, wave tile 32x32 (x2 matrices)
// =====================================================================
constexpr int BM = 128, BN = 64, BK = 32;
constexpr int LDA = BK + 8;   // 40 bf16 -> 80B row stride (16B aligned)
constexpr int LDB = BK + 8;

__global__ __launch_bounds__(256)
void moe_gemm_gateup(const __hip_bfloat16* __restrict__ xb,
                     const float* __restrict__ w_gate,
                     const float* __restrict__ w_up,
                     __hip_bfloat16* __restrict__ hmid) {
  const int e      = blockIdx.z;
  const int mBlock = blockIdx.y * BM;     // in CAP
  const int nBlock = blockIdx.x * BN;     // in F
  const int tid    = threadIdx.x;
  const int lane   = tid & 31;
  const int wave   = tid >> 5;
  const int waveM  = (wave >> 1) * 32;    // 4 waves down
  const int waveN  = (wave & 1) * 32;     // 2 waves across
  const int hi     = (lane >> 4) & 1;
  const int nIdx   = lane & 15;

  __shared__ __align__(16) __hip_bfloat16 As[BM * LDA];
  __shared__ __align__(16) __hip_bfloat16 Bg[BN * LDB];
  __shared__ __align__(16) __hip_bfloat16 Bu[BN * LDB];

  const __hip_bfloat16* arow = xb + (size_t)(e * CAPc + mBlock) * Hc;
  const float* wg = w_gate + (size_t)e * Hc * Fc + nBlock;
  const float* wu = w_up   + (size_t)e * Hc * Fc + nBlock;

  v8f accG[2][2], accU[2][2];
  for (int i = 0; i < 2; ++i)
    for (int j = 0; j < 2; ++j) { accG[i][j] = (v8f)(0.f); accU[i][j] = (v8f)(0.f); }

  // staging indices
  const int ar = tid >> 2;              // A: row 0..63 (+64 second pass)
  const int ag = (tid & 3) * 8;         // A: 16B chunk within row
  const int bk = (tid >> 4) * 2;        // B: even k 0..30
  const int bn = (tid & 15) * 4;        // B: n group of 4

  for (int k0 = 0; k0 < Hc; k0 += BK) {
    // --- stage A: 128x32 bf16 via async global->LDS (ASYNCcnt) ---
    #pragma unroll
    for (int p = 0; p < 2; ++p) {
      const int row = ar + p * 64;
      async_copy_b128(arow + (size_t)row * Hc + k0 + ag, &As[row * LDA + ag]);
    }
    // --- stage B tiles transposed Bt[n][k]: fp32 -> packed bf16x2 b32 stores ---
    {
      const float4 g0 = *(const float4*)(wg + (size_t)(k0 + bk) * Fc + bn);
      const float4 g1 = *(const float4*)(wg + (size_t)(k0 + bk + 1) * Fc + bn);
      const float4 u0 = *(const float4*)(wu + (size_t)(k0 + bk) * Fc + bn);
      const float4 u1 = *(const float4*)(wu + (size_t)(k0 + bk + 1) * Fc + bn);
      *(unsigned*)&Bg[(bn + 0) * LDB + bk] = pack2bf(g0.x, g1.x);
      *(unsigned*)&Bg[(bn + 1) * LDB + bk] = pack2bf(g0.y, g1.y);
      *(unsigned*)&Bg[(bn + 2) * LDB + bk] = pack2bf(g0.z, g1.z);
      *(unsigned*)&Bg[(bn + 3) * LDB + bk] = pack2bf(g0.w, g1.w);
      *(unsigned*)&Bu[(bn + 0) * LDB + bk] = pack2bf(u0.x, u1.x);
      *(unsigned*)&Bu[(bn + 1) * LDB + bk] = pack2bf(u0.y, u1.y);
      *(unsigned*)&Bu[(bn + 2) * LDB + bk] = pack2bf(u0.z, u1.z);
      *(unsigned*)&Bu[(bn + 3) * LDB + bk] = pack2bf(u0.w, u1.w);
    }
    wait_async0();
    __syncthreads();

    const v16bf a0  = load_frag(As, waveM +  0, LDA, lane);
    const v16bf a1  = load_frag(As, waveM + 16, LDA, lane);
    const v16bf bg0 = load_frag(Bg, waveN +  0, LDB, lane);
    const v16bf bg1 = load_frag(Bg, waveN + 16, LDB, lane);
    const v16bf bu0 = load_frag(Bu, waveN +  0, LDB, lane);
    const v16bf bu1 = load_frag(Bu, waveN + 16, LDB, lane);

    accG[0][0] = WMMA_BF16(a0, bg0, accG[0][0]);
    accG[0][1] = WMMA_BF16(a0, bg1, accG[0][1]);
    accG[1][0] = WMMA_BF16(a1, bg0, accG[1][0]);
    accG[1][1] = WMMA_BF16(a1, bg1, accG[1][1]);
    accU[0][0] = WMMA_BF16(a0, bu0, accU[0][0]);
    accU[0][1] = WMMA_BF16(a0, bu1, accU[0][1]);
    accU[1][0] = WMMA_BF16(a1, bu0, accU[1][0]);
    accU[1][1] = WMMA_BF16(a1, bu1, accU[1][1]);
    __syncthreads();
  }

  // --- epilogue: silu(gate) * up -> bf16 hmid (fast hardware rcp) ---
  #pragma unroll
  for (int mt = 0; mt < 2; ++mt) {
    #pragma unroll
    for (int nt = 0; nt < 2; ++nt) {
      const v8f g = accG[mt][nt];
      const v8f u = accU[mt][nt];
      const int baseRow = mBlock + waveM + mt * 16 + 8 * hi;
      const int col     = nBlock + waveN + nt * 16 + nIdx;
      #pragma unroll
      for (int r = 0; r < 8; ++r) {
        const float gv = g[r];
        const float s  = gv * __builtin_amdgcn_rcpf(1.f + __expf(-gv));
        hmid[(size_t)(e * CAPc + baseRow + r) * Fc + col] =
            __float2bfloat16(s * u[r]);
      }
    }
  }
}

// =====================================================================
// 5) GEMM2: yb = hmid @ w_down  per expert  (M=512, N=2048, K=768)
// =====================================================================
__global__ __launch_bounds__(256)
void moe_gemm_down(const __hip_bfloat16* __restrict__ hmid,
                   const float* __restrict__ w_down,
                   __hip_bfloat16* __restrict__ yb) {
  const int e      = blockIdx.z;
  const int mBlock = blockIdx.y * BM;     // in CAP
  const int nBlock = blockIdx.x * BN;     // in H
  const int tid    = threadIdx.x;
  const int lane   = tid & 31;
  const int wave   = tid >> 5;
  const int waveM  = (wave >> 1) * 32;
  const int waveN  = (wave & 1) * 32;
  const int hi     = (lane >> 4) & 1;
  const int nIdx   = lane & 15;

  __shared__ __align__(16) __hip_bfloat16 As[BM * LDA];
  __shared__ __align__(16) __hip_bfloat16 Bt[BN * LDB];

  const __hip_bfloat16* arow = hmid + (size_t)(e * CAPc + mBlock) * Fc;
  const float* wd = w_down + (size_t)e * Fc * Hc + nBlock;

  v8f acc[2][2];
  for (int i = 0; i < 2; ++i)
    for (int j = 0; j < 2; ++j) acc[i][j] = (v8f)(0.f);

  const int ar = tid >> 2;
  const int ag = (tid & 3) * 8;
  const int bk = (tid >> 4) * 2;
  const int bn = (tid & 15) * 4;

  for (int k0 = 0; k0 < Fc; k0 += BK) {
    #pragma unroll
    for (int p = 0; p < 2; ++p) {
      const int row = ar + p * 64;
      async_copy_b128(arow + (size_t)row * Fc + k0 + ag, &As[row * LDA + ag]);
    }
    {
      const float4 v0 = *(const float4*)(wd + (size_t)(k0 + bk) * Hc + bn);
      const float4 v1 = *(const float4*)(wd + (size_t)(k0 + bk + 1) * Hc + bn);
      *(unsigned*)&Bt[(bn + 0) * LDB + bk] = pack2bf(v0.x, v1.x);
      *(unsigned*)&Bt[(bn + 1) * LDB + bk] = pack2bf(v0.y, v1.y);
      *(unsigned*)&Bt[(bn + 2) * LDB + bk] = pack2bf(v0.z, v1.z);
      *(unsigned*)&Bt[(bn + 3) * LDB + bk] = pack2bf(v0.w, v1.w);
    }
    wait_async0();
    __syncthreads();

    const v16bf a0 = load_frag(As, waveM +  0, LDA, lane);
    const v16bf a1 = load_frag(As, waveM + 16, LDA, lane);
    const v16bf b0 = load_frag(Bt, waveN +  0, LDB, lane);
    const v16bf b1 = load_frag(Bt, waveN + 16, LDB, lane);

    acc[0][0] = WMMA_BF16(a0, b0, acc[0][0]);
    acc[0][1] = WMMA_BF16(a0, b1, acc[0][1]);
    acc[1][0] = WMMA_BF16(a1, b0, acc[1][0]);
    acc[1][1] = WMMA_BF16(a1, b1, acc[1][1]);
    __syncthreads();
  }

  #pragma unroll
  for (int mt = 0; mt < 2; ++mt) {
    #pragma unroll
    for (int nt = 0; nt < 2; ++nt) {
      const v8f c = acc[mt][nt];
      const int baseRow = mBlock + waveM + mt * 16 + 8 * hi;
      const int col     = nBlock + waveN + nt * 16 + nIdx;
      #pragma unroll
      for (int r = 0; r < 8; ++r)
        yb[(size_t)(e * CAPc + baseRow + r) * Hc + col] = __float2bfloat16(c[r]);
    }
  }
}

// =====================================================================
// 6) Combine: out[t] = sum_k w[t,k] * yb[dest[t,k]]
// =====================================================================
__global__ __launch_bounds__(256)
void moe_combine(const __hip_bfloat16* __restrict__ yb,
                 const int* __restrict__ dest, const float* __restrict__ topk_w,
                 float* __restrict__ out) {
  const int t  = blockIdx.x;
  const int c0 = threadIdx.x * 8;

  __shared__ int   sd[Kc];
  __shared__ float sw[Kc];
  if (threadIdx.x < Kc) {
    sd[threadIdx.x] = dest[t * Kc + threadIdx.x];
    sw[threadIdx.x] = topk_w[t * Kc + threadIdx.x];
  }
  __syncthreads();

  float acc[8];
  #pragma unroll
  for (int j = 0; j < 8; ++j) acc[j] = 0.f;

  for (int k = 0; k < Kc; ++k) {
    const int d = sd[k];
    if (d < Ec * CAPc) {
      const float w = sw[k];
      union { uint4 q; __hip_bfloat16 h[8]; } pk;
      pk.q = *(const uint4*)(yb + (size_t)d * Hc + c0);
      #pragma unroll
      for (int j = 0; j < 8; ++j) acc[j] += w * __bfloat162float(pk.h[j]);
    }
  }

  float4 o0 = make_float4(acc[0], acc[1], acc[2], acc[3]);
  float4 o1 = make_float4(acc[4], acc[5], acc[6], acc[7]);
  *(float4*)(out + (size_t)t * Hc + c0)     = o0;
  *(float4*)(out + (size_t)t * Hc + c0 + 4) = o1;
}

// =====================================================================
// launch
// =====================================================================
extern "C" void kernel_launch(void* const* d_in, const int* in_sizes, int n_in,
                              void* d_out, int out_size, void* d_ws, size_t ws_size,
                              hipStream_t stream) {
  const float* x  = (const float*)d_in[0];   // hidden_states [B,S,H]
  const float* rw = (const float*)d_in[1];   // router_w [E,H]
  const float* wg = (const float*)d_in[2];   // w_gate [E,H,F]
  const float* wu = (const float*)d_in[3];   // w_up   [E,H,F]
  const float* wd = (const float*)d_in[4];   // w_down [E,F,H]
  float* out = (float*)d_out;

  char* p = (char*)d_ws;
  auto alloc = [&](size_t bytes) {
    char* r = p;
    p += (bytes + 255) & ~(size_t)255;
    return r;
  };
  int*   topk_idx = (int*)  alloc((size_t)Tc * Kc * 4);
  float* topk_w   = (float*)alloc((size_t)Tc * Kc * 4);
  int*   dest     = (int*)  alloc((size_t)Tc * Kc * 4);
  int*   token_of = (int*)  alloc((size_t)Ec * CAPc * 4);
  int*   counts   = (int*)  alloc((size_t)Ec * 4);
  __hip_bfloat16* hmid = (__hip_bfloat16*)alloc((size_t)Ec * CAPc * Fc * 2);
  __hip_bfloat16* xb   = (__hip_bfloat16*)alloc((size_t)Ec * CAPc * Hc * 2);
  __hip_bfloat16* yb   = xb;   // xb is dead after GEMM1; alias to save workspace

  moe_router<<<Tc, 64, 0, stream>>>(x, rw, topk_idx, topk_w);
  moe_dispatch<<<1, 64, 0, stream>>>(topk_idx, dest, token_of, counts);
  moe_gather<<<Ec * CAPc, 256, 0, stream>>>(x, token_of, counts, xb);
  moe_gemm_gateup<<<dim3(Fc / BN, CAPc / BM, Ec), 256, 0, stream>>>(xb, wg, wu, hmid);
  moe_gemm_down<<<dim3(Hc / BN, CAPc / BM, Ec), 256, 0, stream>>>(hmid, wd, yb);
  moe_combine<<<Tc, 256, 0, stream>>>(yb, dest, topk_w, out);
}
